// HybridMemory_62474594288059
// MI455X (gfx1250) — matrix operand order repaired and checked by reading, not compiled, not run on hardware
//
#include <hip/hip_runtime.h>

// ---------------------------------------------------------------------------
// HybridMemory loss on gfx1250:
//   logits = inputs @ cluster_features^T / 0.05        [256 x 131072]
//   S_b    = sum_n exp(logits[b,n])
//   loss   = -mean_b log( exp(logits[b, t_b]) / (S_b + 1e-6) + 1e-6 )
// Split-bf16 (hi+lo) 3x v_wmma_f32_16x16x32_bf16 for ~f32 accuracy at
// matrix-engine rate; cluster_features streamed once (134 MB @ 23.3 TB/s).
// Distance-2 software pipeline: at tile t we issue tile t+2's global loads,
// so the wait for tile t+1's data lands after a full WMMA phase.
// Double-buffered LDS, one barrier per tile.
// ---------------------------------------------------------------------------

typedef __attribute__((ext_vector_type(16))) __bf16 v16bf;
typedef __attribute__((ext_vector_type(4)))  __bf16 v4bf;
typedef __attribute__((ext_vector_type(8)))  float  v8f;

#define BB        256        // batch
#define DD        256        // feature dim (K)
#define NN        131072     // bank size
#define INV_TEMP  20.0f
#define KC        8          // K chunks of 32
#define TPB       512        // 16 waves
#define NT_PER_WG 8          // n-tiles (16 cols) per workgroup

__device__ __forceinline__ v8f wmma_bf16(v16bf a, v16bf b, v8f c) {
    return __builtin_amdgcn_wmma_f32_16x16x32_bf16(
        false, a, false, b, (short)0, c, false, false);
}

// split one float4 into bf16 hi / lo quads, packed stores (ds_store_b64)
__device__ __forceinline__ void cvt_store(__bf16* ph, __bf16* pl, float4 v) {
    __bf16 h0 = (__bf16)v.x, h1 = (__bf16)v.y,
           h2 = (__bf16)v.z, h3 = (__bf16)v.w;
    v4bf hv = {h0, h1, h2, h3};
    v4bf lv = {(__bf16)(v.x - (float)h0), (__bf16)(v.y - (float)h1),
               (__bf16)(v.z - (float)h2), (__bf16)(v.w - (float)h3)};
    *(v4bf*)ph = hv;
    *(v4bf*)pl = lv;
}

// ---------------------------------------------------------------------------
// Kernel A: streaming GEMM + exp + per-row sum accumulation
// ---------------------------------------------------------------------------
__global__ __launch_bounds__(TPB, 1)
void hm_expsum_kernel(const float* __restrict__ inputs,
                      const float* __restrict__ cluster,
                      float* __restrict__ S)
{
    __shared__ __bf16 bh[2][16 * DD];   // cluster tile, bf16 hi (2 x 8 KB)
    __shared__ __bf16 bl[2][16 * DD];   // cluster tile, bf16 lo (2 x 8 KB)

    const int tid    = threadIdx.x;
    const int wave   = tid >> 5;          // 0..15  -> M-tile (16 rows each)
    const int lane   = tid & 31;
    const int lane16 = lane & 15;
    const int hiHalf = (lane >= 16) ? 1 : 0;

    // ---- Resident A fragments: rows wave*16 .. wave*16+15, all K ----------
    // A 16x32 layout (contiguous, matches B): lanes 0-15 hold K kc*32+0..15,
    // lanes 16-31 hold K kc*32+16..31; element e -> K=kb+e.
    v16bf a_hi[KC], a_lo[KC];
    {
        const float* arow = inputs + (size_t)(wave * 16 + lane16) * DD;
        for (int kc = 0; kc < KC; ++kc) {
            const int kb = kc * 32 + (hiHalf ? 16 : 0);
            v16bf h, l;
            for (int j = 0; j < 8; ++j) {
                float x0 = arow[kb + j];
                float x1 = arow[kb + 8 + j];
                __bf16 h0 = (__bf16)x0;
                __bf16 h1 = (__bf16)x1;
                h[j]     = h0;  l[j]     = (__bf16)(x0 - (float)h0);
                h[8 + j] = h1;  l[8 + j] = (__bf16)(x1 - (float)h1);
            }
            a_hi[kc] = h;
            a_lo[kc] = l;
        }
    }

    v8f rowAcc = (v8f)0.0f;   // per-lane partial of sum_n exp(logit)

    const int ntile0 = blockIdx.x * NT_PER_WG;

    // Each thread owns two float4 slots of a 16x256 f32 tile (1024 float4).
    // Distance-2 pipeline: cur = tile t data, nxt = tile t+1 data in flight.
    float4 cur0, cur1, nxt0, nxt1;
    {
        const float4* g0 = (const float4*)(cluster + (size_t)ntile0 * 16 * DD);
        const float4* g1 = (const float4*)(cluster + (size_t)(ntile0 + 1) * 16 * DD);
        cur0 = g0[tid];
        cur1 = g0[tid + TPB];
        nxt0 = g1[tid];
        nxt1 = g1[tid + TPB];
    }

#pragma unroll
    for (int t = 0; t < NT_PER_WG; ++t) {
        const int buf = t & 1;

        // ---- Convert tile t registers -> bf16 hi/lo LDS -------------------
        cvt_store(&bh[buf][4 * tid],           &bl[buf][4 * tid],           cur0);
        cvt_store(&bh[buf][4 * tid + 4 * TPB], &bl[buf][4 * tid + 4 * TPB], cur1);

        // advance pipeline: cur <- tile t+1 (in flight), issue tile t+2
        cur0 = nxt0;
        cur1 = nxt1;
        if (t + 2 < NT_PER_WG) {
            const float4* gp =
                (const float4*)(cluster + (size_t)(ntile0 + t + 2) * 16 * DD);
            nxt0 = gp[tid];
            nxt1 = gp[tid + TPB];
        }

        __syncthreads();   // single barrier per tile (double-buffered LDS)

        // ---- 16x16 logits tile for this wave's 16 rows --------------------
        // B 32x16 bf16 layout: lanes 0-15 hold N=lane16, K=kc*32+0..15;
        // lanes 16-31 hold N=lane16, K=kc*32+16..31 (16 contiguous bf16).
        v8f c0 = (v8f)0.0f;   // hi*hi chain
        v8f c1 = (v8f)0.0f;   // cross-term chain
        for (int kc = 0; kc < KC; ++kc) {
            const int bo = lane16 * DD + kc * 32 + (hiHalf ? 16 : 0);
            v16bf bhf = *(const v16bf*)&bh[buf][bo];
            v16bf blf = *(const v16bf*)&bl[buf][bo];
            c0 = wmma_bf16(a_hi[kc], bhf, c0);   // hi*hi
            c1 = wmma_bf16(a_hi[kc], blf, c1);   // hi*lo
            c1 = wmma_bf16(a_lo[kc], bhf, c1);   // lo*hi
        }

        // exp(logit / TEMP) and accumulate per-row partials (lane = one n)
        for (int r = 0; r < 8; ++r)
            rowAcc[r] += __expf((c0[r] + c1[r]) * INV_TEMP);
    }

    // ---- Reduce over the 16 lanes of each half, then global accumulate ----
    // C layout: VGPR r, lanes 0-15 -> M = wave*16 + r ; lanes 16-31 -> +8.
    for (int r = 0; r < 8; ++r) {
        float v = rowAcc[r];
        v += __shfl_xor(v, 1);
        v += __shfl_xor(v, 2);
        v += __shfl_xor(v, 4);
        v += __shfl_xor(v, 8);
        if (lane16 == 0)
            atomicAdd(&S[wave * 16 + hiHalf * 8 + r], v);
    }
}

// ---------------------------------------------------------------------------
// Kernel B: exact f32 target-column logit  Lt[b] = dot(inputs[b], cluster[t_b]) / TEMP
// ---------------------------------------------------------------------------
__global__ __launch_bounds__(32)
void hm_target_kernel(const float* __restrict__ inputs,
                      const float* __restrict__ cluster,
                      const int* __restrict__ indexes,
                      const int* __restrict__ labels,
                      float* __restrict__ Lt)
{
    const int b    = blockIdx.x;
    const int lane = threadIdx.x;
    const int tgt  = labels[indexes[b]];
    const float* a = inputs + (size_t)b * DD;
    const float* w = cluster + (size_t)tgt * DD;

    float acc = 0.0f;
    for (int d = lane; d < DD; d += 32)
        acc = fmaf(a[d], w[d], acc);
    acc += __shfl_xor(acc, 1);
    acc += __shfl_xor(acc, 2);
    acc += __shfl_xor(acc, 4);
    acc += __shfl_xor(acc, 8);
    acc += __shfl_xor(acc, 16);
    if (lane == 0) Lt[b] = acc * INV_TEMP;
}

// ---------------------------------------------------------------------------
// Kernel C: loss = -mean_b log( exp(Lt[b]) / (S[b] + 1e-6) + 1e-6 )
// ---------------------------------------------------------------------------
__global__ __launch_bounds__(BB)
void hm_finalize_kernel(const float* __restrict__ S,
                        const float* __restrict__ Lt,
                        float* __restrict__ out)
{
    __shared__ float red[BB];
    const int b = threadIdx.x;
    float e    = __expf(Lt[b]);
    float p    = e / (S[b] + 1e-6f);
    red[b]     = -__logf(p + 1e-6f);
    __syncthreads();
    for (int s = BB / 2; s > 0; s >>= 1) {
        if (b < s) red[b] += red[b + s];
        __syncthreads();
    }
    if (b == 0) out[0] = red[0] * (1.0f / BB);
}

// ---------------------------------------------------------------------------
extern "C" void kernel_launch(void* const* d_in, const int* in_sizes, int n_in,
                              void* d_out, int out_size, void* d_ws, size_t ws_size,
                              hipStream_t stream)
{
    const float* inputs  = (const float*)d_in[0];   // [256,256]
    const float* cluster = (const float*)d_in[1];   // [131072,256]
    // d_in[2] = instance_features (unused by reference)
    const int*   indexes = (const int*)d_in[3];     // [256]
    const int*   labels  = (const int*)d_in[4];     // [131072]

    float* S  = (float*)d_ws;        // [256] exp-sum accumulators
    float* Lt = S + BB;              // [256] target logits

    hipMemsetAsync(S, 0, BB * sizeof(float), stream);

    const int NTILES = NN / 16;                    // 8192
    dim3 grid(NTILES / NT_PER_WG);                 // 1024 workgroups
    hm_expsum_kernel<<<grid, TPB, 0, stream>>>(inputs, cluster, S);
    hm_target_kernel<<<BB, 32, 0, stream>>>(inputs, cluster, indexes, labels, Lt);
    hm_finalize_kernel<<<1, BB, 0, stream>>>(S, Lt, (float*)d_out);
}
